// RNN_80032420593910
// MI455X (gfx1250) — compile-verified, matrix operand order checked by reading, not run on hardware
//
#include <hip/hip_runtime.h>
#include <hip/hip_bf16.h>

// ---------------------------------------------------------------------------
// RNN scan for MI455X (gfx1250, wave32, WMMA bf16 16x16x32).
//
//   pack_weights : fp32 -> bf16, WMMA B-operand tile layout, in d_ws (~704 KB).
//   pack_rnnin   : (fast path) precompute rnn_in = x@Wcat^T for ALL timesteps
//                  with 8000 independent WGs (whole GPU busy), stored bf16 in
//                  scan-thread ownership order. Off the scan's critical path.
//   rnn_scan     : 16 persistent WGs x 256 thr (8 waves), WG owns 16 batch rows.
//                  h state in registers (WMMA C-layout) + bf16 A-layout LDS
//                  copy for cross-wave WMMA. W_rec streamed from L2 through a
//                  3-deep rotating operand pipeline (loads issued ~2 k-groups
//                  ahead of the consuming v_wmma). W_out staged in LDS.
//                  2 barriers per step.
// ---------------------------------------------------------------------------

typedef __attribute__((ext_vector_type(16))) __bf16 v16bf;
typedef __attribute__((ext_vector_type(8)))  float  v8f;

#define RULE_START 85
#define N_RULE     20
#define N_IN       105
#define KIN_PAD    128
#define N_OUTPUT   33
#define HIDDEN     512
#define T_STEPS    500
#define BATCH      256
#define ALPHA_     0.2f
#define SIGMA_     0.05f

#define KT_IN   4          // 128/32 k-tiles (input GEMM)
#define KT_H    16         // 512/32 k-tiles (recurrent / output GEMM)
#define NG      (KT_IN + KT_H)   // fused k-group count
#define NT_H    32         // 512/16 n-tiles (hidden)
#define NT_O    3          // 48/16  n-tiles (output, 33 padded to 48)
#define TILE_ELEMS 512     // one packed 32K x 16N bf16 tile: 32 lanes * 16 elems

#define WCAT_ELEMS (KT_IN * NT_H * TILE_ELEMS)   // 65536  (128 KB)
#define WREC_ELEMS (KT_H  * NT_H * TILE_ELEMS)   // 262144 (512 KB)
#define WOUT_ELEMS (KT_H  * NT_O * TILE_ELEMS)   // 24576  (48 KB)
#define WEIGHT_ELEMS (WCAT_ELEMS + WREC_ELEMS + WOUT_ELEMS)
#define PRE_ELEMS  ((size_t)T_STEPS * BATCH * HIDDEN)   // bf16 rnn_in, 131 MB

__device__ __forceinline__ unsigned short f32_to_bf16_rne(float f) {
    unsigned int u = __float_as_uint(f);
    u += 0x7FFFu + ((u >> 16) & 1u);
    return (unsigned short)(u >> 16);
}
__device__ __forceinline__ unsigned int pk_bf16x2(float lo, float hi) {
    return (unsigned int)f32_to_bf16_rne(lo) | ((unsigned int)f32_to_bf16_rne(hi) << 16);
}
// 16-bit operand lane layout inside a 32K x 16 tile:
//   lane = idx16 + 16*(((k&31)>>3)&1),  slot = (k&7) + 8*((k&31)>>4)
__device__ __forceinline__ int pack_lane(int idx16, int kk) {
    return idx16 + ((((kk) >> 3) & 1) << 4);
}
__device__ __forceinline__ int pack_slot(int kk) {
    return (kk & 7) + ((kk >> 4) << 3);
}
__device__ __forceinline__ v16bf ldsv(const unsigned short* p) {
    return *reinterpret_cast<const v16bf*>(p);
}

// ---------------------------------------------------------------------------
// Prep: convert + pack all weights into WMMA B-layout bf16 tiles in ws.
// ---------------------------------------------------------------------------
__global__ void pack_weights(const float* __restrict__ W_sens,
                             const float* __restrict__ W_rule,
                             const float* __restrict__ W_rec,
                             const float* __restrict__ W_out,
                             unsigned short* __restrict__ ws) {
    int idx = blockIdx.x * blockDim.x + threadIdx.x;
    if (idx >= WEIGHT_ELEMS) return;

    int which, local, ntiles;
    unsigned short* dst;
    if (idx < WCAT_ELEMS)                 { which = 0; local = idx;                          ntiles = NT_H; dst = ws; }
    else if (idx < WCAT_ELEMS+WREC_ELEMS) { which = 1; local = idx - WCAT_ELEMS;            ntiles = NT_H; dst = ws + WCAT_ELEMS; }
    else                                  { which = 2; local = idx - WCAT_ELEMS-WREC_ELEMS; ntiles = NT_O; dst = ws + WCAT_ELEMS + WREC_ELEMS; }

    int tile = local >> 9;
    int rem  = local & 511;
    int lane = rem >> 4;
    int slot = rem & 15;
    int nt = tile % ntiles;
    int kt = tile / ntiles;
    int kk = (slot & 7) | (((lane >> 4) & 1) << 3) | ((slot >> 3) << 4);
    int k  = kt * 32 + kk;
    int n  = nt * 16 + (lane & 15);

    float v = 0.f;
    if (which == 0) {
        if (k < RULE_START)      v = W_sens[n * RULE_START + k];
        else if (k < N_IN)       v = W_rule[n * N_RULE + (k - RULE_START)];
    } else if (which == 1) {
        v = W_rec[n * HIDDEN + k];
    } else {
        if (n < N_OUTPUT)        v = W_out[n * HIDDEN + k];
    }
    dst[local] = f32_to_bf16_rne(v);
}

// ---------------------------------------------------------------------------
// Fast path pre-pass: rnn_in[t] = x_t @ Wcat^T for all t, fully parallel.
// Output layout matches scan thread ownership: [t][wg][tid][j*8+jj] bf16.
// ---------------------------------------------------------------------------
__global__ __launch_bounds__(256, 1)
void pack_rnnin(const float* __restrict__ x,
                const unsigned short* __restrict__ ws,
                unsigned short* __restrict__ pre) {
    __shared__ __align__(16) unsigned short AX[KT_IN * TILE_ELEMS];
    const int tb = blockIdx.x;
    const int t  = tb >> 4;
    const int wg = tb & 15;
    const int tid = threadIdx.x, lane = tid & 31, w = tid >> 5, w4 = w * 4;
    const int bbase = wg * 16;
    const unsigned short* WcatP = ws;

    for (int i = tid; i < 16 * KIN_PAD; i += 256) {
        int r = i >> 7, f = i & 127;
        float v = (f < N_IN) ? x[((size_t)t * BATCH + bbase + r) * N_IN + f] : 0.f;
        int kt = f >> 5, kk = f & 31;
        AX[(kt * 32 + pack_lane(r, kk)) * 16 + pack_slot(kk)] = f32_to_bf16_rne(v);
    }
    __syncthreads();

    v8f acc[4];
    acc[0] = {}; acc[1] = {}; acc[2] = {}; acc[3] = {};
    #pragma unroll
    for (int kt = 0; kt < KT_IN; ++kt) {
        v16bf a = ldsv(&AX[(kt * 32 + lane) * 16]);
        const unsigned short* bp = WcatP + ((size_t)(kt * NT_H + w4)) * TILE_ELEMS + lane * 16;
        #pragma unroll
        for (int j = 0; j < 4; ++j) {
            v16bf b = *reinterpret_cast<const v16bf*>(bp + (size_t)j * TILE_ELEMS);
            acc[j] = __builtin_amdgcn_wmma_f32_16x16x32_bf16(
                false, a, false, b, (short)0, acc[j], false, false);
        }
    }

    uint4* dst = reinterpret_cast<uint4*>(pre) + (((size_t)t * 16 + wg) * 256 + tid) * 4;
    #pragma unroll
    for (int j = 0; j < 4; ++j) {
        uint4 q;
        q.x = pk_bf16x2(acc[j][0], acc[j][1]);
        q.y = pk_bf16x2(acc[j][2], acc[j][3]);
        q.z = pk_bf16x2(acc[j][4], acc[j][5]);
        q.w = pk_bf16x2(acc[j][6], acc[j][7]);
        dst[j] = q;
    }
}

// ---------------------------------------------------------------------------
// Persistent fused scan. USE_PRE: input GEMM precomputed (fast path).
// ---------------------------------------------------------------------------
template <bool USE_PRE>
__global__ __launch_bounds__(256, 1)
void rnn_scan(const float* __restrict__ x,
              const float* __restrict__ noise,
              const float* __restrict__ b_sens,
              const float* __restrict__ b_rec,
              const float* __restrict__ b_out,
              const unsigned short* __restrict__ ws,
              const unsigned short* __restrict__ rnnpre,
              float* __restrict__ out) {
    // Hot A-operand array first => ds immediate offsets (<64KB).
    __shared__ __align__(16) unsigned short ApackH[KT_H * TILE_ELEMS];          // 16 KB
    __shared__ __align__(16) unsigned short WoutLDS[WOUT_ELEMS];                // 48 KB
    __shared__ __align__(16) unsigned short ApackX[USE_PRE ? 8 : KT_IN * TILE_ELEMS];
    __shared__ __align__(16) unsigned short WcatLDS[USE_PRE ? 8 : WCAT_ELEMS];

    const unsigned short* WrecP = ws + WCAT_ELEMS;
    const unsigned short* WoutP = ws + WCAT_ELEMS + WREC_ELEMS;

    const int tid   = threadIdx.x;
    const int lane  = tid & 31;
    const int w     = tid >> 5;
    const int w4    = w * 4;
    const int wg    = blockIdx.x;
    const int bbase = wg * 16;
    const int mrow  = ((lane >> 4) << 3);   // C/D layout: lanes>=16 hold M+8

    // ---- one-time init ------------------------------------------------------
    {
        const uint4* s1 = reinterpret_cast<const uint4*>(WoutP);
        uint4*       d1 = reinterpret_cast<uint4*>(WoutLDS);
        for (int i = tid; i < WOUT_ELEMS / 8; i += 256) d1[i] = s1[i];
    }
    if constexpr (!USE_PRE) {
        const uint4* s0 = reinterpret_cast<const uint4*>(ws);
        uint4*       d0 = reinterpret_cast<uint4*>(WcatLDS);
        for (int i = tid; i < WCAT_ELEMS / 8; i += 256) d0[i] = s0[i];
        for (int i = tid; i < 16 * KIN_PAD; i += 256) {   // pack x_0
            int r = i >> 7, f = i & 127;
            float v = (f < N_IN) ? x[(size_t)(bbase + r) * N_IN + f] : 0.f;
            int kt = f >> 5, kk = f & 31;
            ApackX[(kt * 32 + pack_lane(r, kk)) * 16 + pack_slot(kk)] = f32_to_bf16_rne(v);
        }
    }
    for (int i = tid; i < KT_H * TILE_ELEMS; i += 256) ApackH[i] = 0;

    v8f hreg[4];
    hreg[0] = {}; hreg[1] = {}; hreg[2] = {}; hreg[3] = {};

    float bias_j[4];
    #pragma unroll
    for (int j = 0; j < 4; ++j) {
        int n = (w4 + j) * 16 + (lane & 15);
        bias_j[j] = b_sens[n] + b_rec[n];
    }
    const int   n_out  = w * 16 + (lane & 15);
    const float bo_out = (w < NT_O && n_out < N_OUTPUT) ? b_out[n_out] : 0.f;

    __syncthreads();

    for (int t = 0; t < T_STEPS; ++t) {
        // ---- Phase 2 --------------------------------------------------------
        // Fast path: issue the precomputed rnn_in loads now (consumed only in
        // phase 3 => fully overlapped with the recurrent WMMAs below).
        uint4 rq[4];
        if constexpr (USE_PRE) {
            const uint4* rp = reinterpret_cast<const uint4*>(rnnpre)
                            + (((size_t)t * 16 + wg) * 256 + tid) * 4;
            #pragma unroll
            for (int j = 0; j < 4; ++j) rq[j] = rp[j];
        }

        v8f acc[4];
        acc[0] = {}; acc[1] = {}; acc[2] = {}; acc[3] = {};
        {
            constexpr int G0    = USE_PRE ? KT_IN : 0;   // skip input groups
            constexpr int DEPTH = USE_PRE ? 3 : 2;       // operand pipeline depth

            auto load_group = [&](int g, v16bf& a, v16bf (&b)[4]) {
                if (g < KT_IN) {             // only reachable when !USE_PRE
                    a = ldsv(&ApackX[(g * 32 + lane) * 16]);
                    #pragma unroll
                    for (int j = 0; j < 4; ++j)
                        b[j] = ldsv(&WcatLDS[(g * NT_H + w4 + j) * TILE_ELEMS + lane * 16]);
                } else {
                    const int kt = g - KT_IN;
                    a = ldsv(&ApackH[(kt * 32 + lane) * 16]);
                    const unsigned short* bp =
                        WrecP + ((size_t)(kt * NT_H + w4)) * TILE_ELEMS + lane * 16;
                    #pragma unroll
                    for (int j = 0; j < 4; ++j)
                        b[j] = *reinterpret_cast<const v16bf*>(bp + (size_t)j * TILE_ELEMS);
                }
            };

            v16bf aa[DEPTH];
            v16bf bb[DEPTH][4];
            #pragma unroll
            for (int d = 0; d < DEPTH; ++d) load_group(G0 + d, aa[d], bb[d]);

            #pragma unroll
            for (int g = G0; g < NG; ++g) {
                const int s = (g - G0) % DEPTH;
                #pragma unroll
                for (int j = 0; j < 4; ++j)
                    acc[j] = __builtin_amdgcn_wmma_f32_16x16x32_bf16(
                        false, aa[s], false, bb[s][j], (short)0, acc[j], false, false);
                if (g + DEPTH < NG) load_group(g + DEPTH, aa[s], bb[s]);
            }
        }
        __syncthreads();   // all ApackH reads done before rewrite

        // ---- Phase 3: rnn_in + bias + sigma*noise, softplus, leaky update ---
        const float* nz = noise + ((size_t)t * BATCH + bbase) * HIDDEN;
        #pragma unroll
        for (int j = 0; j < 4; ++j) {
            int n = (w4 + j) * 16 + (lane & 15);
            int kt = n >> 5, kk = n & 31;
            int slot = pack_slot(kk);
            float rin[8];
            if constexpr (USE_PRE) {
                unsigned int qq[4] = {rq[j].x, rq[j].y, rq[j].z, rq[j].w};
                #pragma unroll
                for (int c = 0; c < 4; ++c) {
                    rin[2*c]   = __uint_as_float(qq[c] << 16);
                    rin[2*c+1] = __uint_as_float(qq[c] & 0xffff0000u);
                }
            } else {
                #pragma unroll
                for (int c = 0; c < 8; ++c) rin[c] = 0.f;
            }
            #pragma unroll
            for (int jj = 0; jj < 8; ++jj) {
                int m = mrow + jj;
                float pre = acc[j][jj] + rin[jj] + bias_j[j]
                          + SIGMA_ * nz[(size_t)m * HIDDEN + n];
                float sp  = fmaxf(pre, 0.f) + __logf(1.f + __expf(-fabsf(pre)));
                float hn  = ALPHA_ * sp + (1.f - ALPHA_) * hreg[j][jj];
                hreg[j][jj] = hn;
                ApackH[(kt * 32 + pack_lane(m, kk)) * 16 + slot] = f32_to_bf16_rne(hn);
            }
        }
        if constexpr (!USE_PRE) {   // pack x_{t+1}
            if (t + 1 < T_STEPS) {
                for (int i = tid; i < 16 * KIN_PAD; i += 256) {
                    int r = i >> 7, f = i & 127;
                    float v = (f < N_IN) ? x[((size_t)(t + 1) * BATCH + bbase + r) * N_IN + f] : 0.f;
                    int kt = f >> 5, kk = f & 31;
                    ApackX[(kt * 32 + pack_lane(r, kk)) * 16 + pack_slot(kk)] = f32_to_bf16_rne(v);
                }
            }
        }
        if (t + 2 < T_STEPS) {   // cache hints for the step after next
            const float* nn = noise + ((size_t)(t + 2) * BATCH + bbase) * HIDDEN;
            __builtin_prefetch(nn + tid * 32, 0, 1);
            if constexpr (USE_PRE) {
                const unsigned short* pp = rnnpre + ((size_t)(t + 2) * 16 + wg) * 256 * 32;
                __builtin_prefetch(pp + tid * 32, 0, 1);
            } else {
                const float* xn = x + ((size_t)(t + 2) * BATCH + bbase) * N_IN;
                __builtin_prefetch(xn + tid * 7, 0, 1);
            }
        }
        __syncthreads();   // h_new packed (and x_{t+1} packed on slow path)

        // ---- Phase 4: y_t = h_new @ W_out^T + b_out (waves 0..2) ------------
        if (w < NT_O) {
            v8f oacc = {};
            v16bf a_cur = ldsv(&ApackH[lane * 16]);
            v16bf b_cur = ldsv(&WoutLDS[w * TILE_ELEMS + lane * 16]);
            #pragma unroll
            for (int kt = 0; kt < KT_H; ++kt) {
                v16bf a_nxt = a_cur, b_nxt = b_cur;
                if (kt + 1 < KT_H) {
                    a_nxt = ldsv(&ApackH[((kt + 1) * 32 + lane) * 16]);
                    b_nxt = ldsv(&WoutLDS[((kt + 1) * NT_O + w) * TILE_ELEMS + lane * 16]);
                }
                oacc = __builtin_amdgcn_wmma_f32_16x16x32_bf16(
                    false, a_cur, false, b_cur, (short)0, oacc, false, false);
                a_cur = a_nxt; b_cur = b_nxt;
            }
            if (n_out < N_OUTPUT) {
                #pragma unroll
                for (int jj = 0; jj < 8; ++jj) {
                    int m = mrow + jj;
                    out[((size_t)t * BATCH + bbase + m) * N_OUTPUT + n_out] = oacc[jj] + bo_out;
                }
            }
        }
        // no barrier: phase 4 and next phase 2 only read; the next write to
        // ApackH (phase 3 of t+1) is gated by the phase-2-end barrier.
    }
}

// ---------------------------------------------------------------------------
extern "C" void kernel_launch(void* const* d_in, const int* in_sizes, int n_in,
                              void* d_out, int out_size, void* d_ws, size_t ws_size,
                              hipStream_t stream) {
    const float* x      = (const float*)d_in[0];
    const float* noise  = (const float*)d_in[1];
    const float* W_sens = (const float*)d_in[2];
    const float* b_sens = (const float*)d_in[3];
    const float* W_rule = (const float*)d_in[4];
    const float* W_rec  = (const float*)d_in[5];
    const float* b_rec  = (const float*)d_in[6];
    const float* W_out  = (const float*)d_in[7];
    const float* b_out  = (const float*)d_in[8];
    float* out          = (float*)d_out;
    unsigned short* ws  = (unsigned short*)d_ws;

    pack_weights<<<(WEIGHT_ELEMS + 255) / 256, 256, 0, stream>>>(W_sens, W_rule, W_rec, W_out, ws);

    const size_t need_fast = ((size_t)WEIGHT_ELEMS + PRE_ELEMS) * sizeof(unsigned short);
    if (ws_size >= need_fast) {
        unsigned short* rnnpre = ws + WEIGHT_ELEMS;
        pack_rnnin<<<T_STEPS * 16, 256, 0, stream>>>(x, ws, rnnpre);
        rnn_scan<true><<<BATCH / 16, 256, 0, stream>>>(
            x, noise, b_sens, b_rec, b_out, ws, rnnpre, out);
    } else {
        rnn_scan<false><<<BATCH / 16, 256, 0, stream>>>(
            x, noise, b_sens, b_rec, b_out, ws, ws, out);
    }
}
